// MTCity_GC_GRU_57088705298496
// MI455X (gfx1250) — matrix operand order repaired and checked by reading, not compiled, not run on hardware
//
#include <hip/hip_runtime.h>

typedef __attribute__((ext_vector_type(16))) _Float16 v16h;
typedef __attribute__((ext_vector_type(8)))  _Float16 v8h;
typedef __attribute__((ext_vector_type(8)))  float    v8f;

#define DEV __device__ __forceinline__

// ---------------------------------------------------------------------------
// WMMA fragment helpers (operand layouts this code uses with v_wmma_f32_16x16x32_f16; A/B k order is contiguous, wave32)
// A (16x32 f16, row-major source): lane = 16*khalf + M ; element e -> K =
// khalf*16+e.  Two b128 loads.
// B (32x16 f16, packed source): lane = 16*kgroup + N ; element e -> K =
// kgroup*16+e.  Pre-packed so each lane reads 32 contiguous bytes.
// C/D (16x16 f32): lane%16 = N col, element r -> row r + 8*(lane/16).
// ---------------------------------------------------------------------------
DEV v16h frag_a(const _Float16* __restrict__ A, int lda, int m0, int kb, int lane) {
  int row = m0 + (lane & 15);
  int kh  = lane >> 4;
  const _Float16* base = A + (size_t)row * lda + kb + kh * 16;
  v8h lo = *(const v8h*)(base);
  v8h hi = *(const v8h*)(base + 8);
  v16h r;
#pragma unroll
  for (int i = 0; i < 8; ++i) { r[i] = lo[i]; r[i + 8] = hi[i]; }
  return r;
}

DEV v16h frag_b(const _Float16* __restrict__ Bp, int tile, int lane) {
  const _Float16* p = Bp + ((size_t)tile * 32 + lane) * 16;
  v8h lo = *(const v8h*)p;
  v8h hi = *(const v8h*)(p + 8);
  v16h r;
#pragma unroll
  for (int i = 0; i < 8; ++i) { r[i] = lo[i]; r[i + 8] = hi[i]; }
  return r;
}

// Fast transcendentals: single v_exp_f32 + v_rcp_f32, no IEEE divide chain.
DEV float sigm(float x) { return __builtin_amdgcn_rcpf(1.f + __expf(-x)); }
DEV float tanh_fast(float x) {
  // tanh(x) = 1 - 2/(exp(2x)+1)
  return 1.f - 2.f * __builtin_amdgcn_rcpf(__expf(2.f * x) + 1.f);
}

// ---------------------------------------------------------------------------
// Generic WMMA GEMM: Out[M,N] = epilogue(A[M,K] @ B[K,N] (+bias)(+Add))
// EPI 0: plain -> f16   EPI 2: +bias +Add(f16) -> f16 at column offset
// EPI 3: +bias -> f32
// Each wave owns a 16 x (16*NT) strip of D: the A fragment is loaded once per
// k-step and reused across NT B fragments / NT back-to-back WMMAs (hides XDL
// latency, cuts A traffic NT-fold).  Block = 256 threads = 8 waves.
// blockIdx.z = batch (strides sA/sB/sAdd/sOut in elements, 0 = shared).
// ---------------------------------------------------------------------------
template <int EPI, int NT>
__global__ void gemm_wmma(const _Float16* __restrict__ A, long long sA, int lda,
                          const _Float16* __restrict__ Bp, long long sB,
                          const float* __restrict__ bias,
                          const _Float16* __restrict__ Add, long long sAdd, int ldadd,
                          void* __restrict__ Out, long long sOut, int ldout, int colOff,
                          int M, int N, int K) {
  const int lane  = threadIdx.x & 31;
  const int wave  = threadIdx.x >> 5;
  const int mtile = blockIdx.x * (blockDim.x >> 5) + wave;
  const int nb    = blockIdx.y * NT;          // first n-tile of this wave
  const int b     = blockIdx.z;
  if (mtile * 16 >= M) return;
  const _Float16* Ab = A + (size_t)b * sA;
  const _Float16* Bb = Bp + (size_t)b * sB;
  const int ntiles = N >> 4;
  const int m0 = mtile * 16;
  v8f c[NT];
#pragma unroll
  for (int j = 0; j < NT; ++j) c[j] = v8f{};
  for (int kb = 0; kb < K; kb += 32) {
    v16h a = frag_a(Ab, lda, m0, kb, lane);
    const int tb = (kb >> 5) * ntiles + nb;
#pragma unroll
    for (int j = 0; j < NT; ++j) {
      v16h bf = frag_b(Bb, tb + j, lane);
      c[j] = __builtin_amdgcn_wmma_f32_16x16x32_f16(false, a, false, bf, (short)0,
                                                    c[j], false, false);
    }
  }
  const int r0 = m0 + ((lane >> 4) << 3);
#pragma unroll
  for (int j = 0; j < NT; ++j) {
    const int col = (nb + j) * 16 + (lane & 15);
    const float bv = (EPI == 0) ? 0.f : bias[col];
#pragma unroll
    for (int r = 0; r < 8; ++r) {
      float v = c[j][r] + bv;
      if (EPI == 2) v += (float)Add[(size_t)b * sAdd + (size_t)(r0 + r) * ldadd + col];
      size_t oi = (size_t)b * sOut + (size_t)(r0 + r) * ldout + colOff + col;
      if (EPI == 3) ((float*)Out)[oi] = v;
      else          ((_Float16*)Out)[oi] = (_Float16)v;
    }
  }
}

// Chebyshev K=2 conv: Out = sigmoid(A1@B1 + A2@B2 + bias).  K=32 per pair,
// N=64 handled as 4 n-tiles per wave with both A fragments reused.
__global__ void cheb_wmma(const _Float16* __restrict__ A1, int lda1,
                          const _Float16* __restrict__ B1,
                          const _Float16* __restrict__ A2, int lda2,
                          const _Float16* __restrict__ B2,
                          const float* __restrict__ bias,
                          _Float16* __restrict__ Out, int ldout, int M) {
  const int lane  = threadIdx.x & 31;
  const int wave  = threadIdx.x >> 5;
  const int mtile = blockIdx.x * (blockDim.x >> 5) + wave;
  if (mtile * 16 >= M) return;
  const int m0 = mtile * 16;
  v8f c[4];
#pragma unroll
  for (int j = 0; j < 4; ++j) c[j] = v8f{};
  {
    v16h a = frag_a(A1, lda1, m0, 0, lane);
#pragma unroll
    for (int j = 0; j < 4; ++j) {
      v16h b = frag_b(B1, j, lane);
      c[j] = __builtin_amdgcn_wmma_f32_16x16x32_f16(false, a, false, b, (short)0,
                                                    c[j], false, false);
    }
  }
  {
    v16h a = frag_a(A2, lda2, m0, 0, lane);
#pragma unroll
    for (int j = 0; j < 4; ++j) {
      v16h b = frag_b(B2, j, lane);
      c[j] = __builtin_amdgcn_wmma_f32_16x16x32_f16(false, a, false, b, (short)0,
                                                    c[j], false, false);
    }
  }
  const int r0 = m0 + ((lane >> 4) << 3);
#pragma unroll
  for (int j = 0; j < 4; ++j) {
    const int col = j * 16 + (lane & 15);
    const float bv = bias[col];
#pragma unroll
    for (int r = 0; r < 8; ++r)
      Out[(size_t)(r0 + r) * ldout + col] = (_Float16)sigm(c[j][r] + bv);
  }
}

// ---------------------------------------------------------------------------
// Packing / setup kernels
// ---------------------------------------------------------------------------
__global__ void pack_w(const float* __restrict__ in, _Float16* __restrict__ out,
                       int Ksrc, int ntiles, int nTilesTotal, int ldin) {
  int tid  = blockIdx.x * blockDim.x + threadIdx.x;
  int lane = tid & 31;
  int tile = tid >> 5;
  if (tile >= nTilesTotal) return;
  int ktile = tile / ntiles, ntile = tile % ntiles;
  int col = ntile * 16 + (lane & 15);
  int kg  = lane >> 4;
  alignas(16) _Float16 buf[16];
#pragma unroll
  for (int e = 0; e < 16; ++e) {
    int k = ktile * 32 + kg * 16 + e;
    buf[e] = (k < Ksrc) ? (_Float16)in[(size_t)k * ldin + col] : (_Float16)0.f;
  }
  _Float16* o = out + ((size_t)tile * 32 + lane) * 16;
  *(v8h*)o = *(const v8h*)buf;
  *(v8h*)(o + 8) = *(const v8h*)(buf + 8);
}

__global__ void pack_b16(const _Float16* __restrict__ in, long long sIn, int ldin,
                         _Float16* __restrict__ out, long long sOut,
                         int ntiles, int nTilesTotal) {
  int tid  = blockIdx.x * blockDim.x + threadIdx.x;
  int lane = tid & 31;
  int tile = tid >> 5;
  if (tile >= nTilesTotal) return;
  int b = blockIdx.z;
  const _Float16* ib = in + (size_t)b * sIn;
  int ktile = tile / ntiles, ntile = tile % ntiles;
  int col = ntile * 16 + (lane & 15);
  int kg  = lane >> 4;
  alignas(16) _Float16 buf[16];
#pragma unroll
  for (int e = 0; e < 16; ++e) {
    int k = ktile * 32 + kg * 16 + e;
    buf[e] = ib[(size_t)k * ldin + col];
  }
  _Float16* o = out + (size_t)b * sOut + ((size_t)tile * 32 + lane) * 16;
  *(v8h*)o = *(const v8h*)buf;
  *(v8h*)(o + 8) = *(const v8h*)(buf + 8);
}

__global__ void deg_dinv(const float* __restrict__ adj, float* __restrict__ dinv, int Nn) {
  int m = blockIdx.x * blockDim.x + threadIdx.x;
  if (m >= Nn) return;
  float s = 0.f;
  for (int n = 0; n < Nn; ++n)
    s += (adj[(size_t)m * Nn + n] != 0.f && n != m) ? 1.f : 0.f;
  dinv[m] = (s > 0.f) ? rsqrtf(s) : 0.f;
}

__global__ void lap_build(const float* __restrict__ adj, const float* __restrict__ dinv,
                          _Float16* __restrict__ L, int Nn) {
  size_t tid = (size_t)blockIdx.x * blockDim.x + threadIdx.x;
  if (tid >= (size_t)Nn * Nn) return;
  int m = (int)(tid / Nn), n = (int)(tid % Nn);
  float a = (adj[tid] != 0.f && m != n) ? 1.f : 0.f;
  L[tid] = (_Float16)(-(dinv[m] * a * dinv[n]));
}

__global__ void afc_conv(const float* __restrict__ afc, _Float16* __restrict__ a16,
                         _Float16* __restrict__ aT16, int Ns, int Nc) {
  size_t tid = (size_t)blockIdx.x * blockDim.x + threadIdx.x;
  if (tid >= (size_t)Ns * Nc) return;
  int s = (int)(tid / Nc), cc = (int)(tid % Nc);
  _Float16 v = (_Float16)afc[tid];
  a16[tid] = v;
  aT16[(size_t)cc * Ns + s] = v;
}

__global__ void init_yprev(const float* __restrict__ x_hist, float* __restrict__ y_buf,
                           int Nn, int Th, int Mrows) {
  int tid = blockIdx.x * blockDim.x + threadIdx.x;
  if (tid >= Mrows) return;
  int b = tid / Nn, n = tid % Nn;
  y_buf[tid] = x_hist[((size_t)b * Th + (Th - 1)) * Nn + n];
}

// Build x_t: Xrow (M,32) f16, Xpack (B-operand layout for L@X), Xgru cols 0..27
// and zero pad cols 92..95.  col0 = x_hist[.,t] (hist) or y_prev (pred),
// cols 1..27 = features[., f_t], cols 28..31 = 0.
__global__ void build_x(const float* __restrict__ x_hist, const float* __restrict__ feats,
                        const float* __restrict__ y_buf,
                        _Float16* __restrict__ Xrow, _Float16* __restrict__ Xpack,
                        _Float16* __restrict__ Xgru,
                        int Nn, int t_hist, int f_t, int use_y, int T, int Dm, int Th,
                        int total) {
  int tid = blockIdx.x * blockDim.x + threadIdx.x;
  if (tid >= total) return;
  int col = tid & 31;
  int n   = (tid >> 5) % Nn;
  int b   = tid / (32 * Nn);
  float v;
  if (col == 0)
    v = use_y ? y_buf[(size_t)b * Nn + n]
              : x_hist[((size_t)b * Th + t_hist) * Nn + n];
  else if (col < 28)
    v = feats[(((size_t)b * T + f_t) * Nn + n) * Dm + (col - 1)];
  else
    v = 0.f;
  _Float16 h = (_Float16)v;
  size_t row = (size_t)b * Nn + n;
  Xrow[row * 32 + col] = h;
  if (col < 28) Xgru[row * 96 + col] = h;
  else          Xgru[row * 96 + 64 + col] = (_Float16)0.f;  // cols 92..95
  // B-operand pack for T1 = L @ X  (K = node dim, N = 32 features)
  int ktile = n >> 5, kg = (n >> 4) & 1, e = n & 15;
  int ntile = col >> 4;
  size_t pidx = (size_t)b * ((size_t)Nn * 32) +
                (((size_t)ktile * 2 + ntile) * 32 + (kg * 16 + (col & 15))) * 16 + e;
  Xpack[pidx] = h;
}

// Multi-timescale GRU update (INTERVALS = 1,2,4,8 over 4 groups of 16 dims).
__global__ void gru_update(const float* __restrict__ Gi, const float* __restrict__ Gh,
                           float* __restrict__ H, _Float16* __restrict__ H16,
                           int Mrows, int idx) {
  int tid = blockIdx.x * blockDim.x + threadIdx.x;
  if (tid >= Mrows * 64) return;
  int m = tid >> 6, j = tid & 63;
  const float* gi = Gi + (size_t)m * 192;
  const float* gh = Gh + (size_t)m * 192;
  float h = H[tid];
  float r = sigm(gi[j] + gh[j]);
  float z = sigm(gi[64 + j] + gh[64 + j]);
  float n = tanh_fast(gi[128 + j] + r * gh[128 + j]);
  float hn = (1.f - z) * n + z * h;
  int interval = 1 << (j >> 4);
  if ((idx & (interval - 1)) != 0) hn = h;
  H[tid]  = hn;
  H16[tid] = (_Float16)hn;
}

__global__ void fc_out(const float* __restrict__ H, const float* __restrict__ fcw,
                       const float* __restrict__ fcb, float* __restrict__ y_buf,
                       float* __restrict__ out, int Nn, int t, int Tp, int Mrows) {
  int tid = blockIdx.x * blockDim.x + threadIdx.x;
  if (tid >= Mrows) return;
  int b = tid / Nn, n = tid % Nn;
  float acc = fcb[0];
  const float* h = H + (size_t)tid * 64;
#pragma unroll
  for (int j = 0; j < 64; ++j) acc += h[j] * fcw[j];
  y_buf[tid] = acc;
  out[((size_t)b * Tp + t) * Nn + n] = acc;
}

// ---------------------------------------------------------------------------
extern "C" void kernel_launch(void* const* d_in, const int* in_sizes, int n_in,
                              void* d_out, int out_size, void* d_ws, size_t ws_size,
                              hipStream_t stream) {
  (void)in_sizes; (void)n_in; (void)out_size; (void)ws_size;
  const int B = 8, Ns = 4096, Nc = 256, Th = 8, Tp = 24, T = 32, Dm = 27;
  const int Ms = B * Ns, Mc = B * Nc;

  const float* x_hist   = (const float*)d_in[0];
  const float* features = (const float*)d_in[1];
  const float* cx_hist  = (const float*)d_in[2];
  const float* cfeats   = (const float*)d_in[3];
  const float* adj_sta  = (const float*)d_in[4];
  const float* adj_city = (const float*)d_in[5];
  const float* afc      = (const float*)d_in[6];
  const float* conv_w   = (const float*)d_in[7];
  const float* conv_b   = (const float*)d_in[8];
  const float* cconv_w  = (const float*)d_in[9];
  const float* cconv_b  = (const float*)d_in[10];
  const float* gru_wi   = (const float*)d_in[11];
  const float* gru_wh   = (const float*)d_in[12];
  const float* gru_bi   = (const float*)d_in[13];
  const float* gru_bh   = (const float*)d_in[14];
  const float* cgru_wi  = (const float*)d_in[15];
  const float* cgru_wh  = (const float*)d_in[16];
  const float* cgru_bi  = (const float*)d_in[17];
  const float* cgru_bh  = (const float*)d_in[18];
  const float* fc_w     = (const float*)d_in[19];
  const float* fc_b     = (const float*)d_in[20];
  const float* cfc_w    = (const float*)d_in[21];
  const float* cfc_b    = (const float*)d_in[22];
  const float* c2s_w    = (const float*)d_in[23];
  const float* c2s_b    = (const float*)d_in[24];
  const float* s2c_w    = (const float*)d_in[25];
  const float* s2c_b    = (const float*)d_in[26];

  float* out_s = (float*)d_out;
  float* out_c = out_s + (size_t)B * Tp * Ns;

  size_t off = 0;
  auto alloc = [&](size_t bytes) -> void* {
    void* p = (char*)d_ws + off;
    off += (bytes + 255) & ~(size_t)255;
    return p;
  };
  _Float16* Ls     = (_Float16*)alloc((size_t)Ns * Ns * 2);
  _Float16* Lc     = (_Float16*)alloc((size_t)Nc * Nc * 2);
  float*    dinv_s = (float*)alloc(Ns * 4);
  float*    dinv_c = (float*)alloc(Nc * 4);
  _Float16* afc16  = (_Float16*)alloc((size_t)Ns * Nc * 2);
  _Float16* afcT16 = (_Float16*)alloc((size_t)Ns * Nc * 2);
  _Float16* W0sp = (_Float16*)alloc(32 * 64 * 2);
  _Float16* W1sp = (_Float16*)alloc(32 * 64 * 2);
  _Float16* W0cp = (_Float16*)alloc(32 * 64 * 2);
  _Float16* W1cp = (_Float16*)alloc(32 * 64 * 2);
  _Float16* wisp = (_Float16*)alloc(96 * 192 * 2);
  _Float16* whsp = (_Float16*)alloc(64 * 192 * 2);
  _Float16* wicp = (_Float16*)alloc(96 * 192 * 2);
  _Float16* whcp = (_Float16*)alloc(64 * 192 * 2);
  _Float16* c2sp = (_Float16*)alloc(64 * 64 * 2);
  _Float16* s2cp = (_Float16*)alloc(64 * 64 * 2);
  _Float16* Xrow_s  = (_Float16*)alloc((size_t)Ms * 32 * 2);
  _Float16* Xpack_s = (_Float16*)alloc((size_t)Ms * 32 * 2);
  _Float16* Xrow_c  = (_Float16*)alloc((size_t)Mc * 32 * 2);
  _Float16* Xpack_c = (_Float16*)alloc((size_t)Mc * 32 * 2);
  _Float16* T1s = (_Float16*)alloc((size_t)Ms * 32 * 2);
  _Float16* T1c = (_Float16*)alloc((size_t)Mc * 32 * 2);
  _Float16* xg_row   = (_Float16*)alloc((size_t)Ms * 64 * 2);
  _Float16* xg_pack  = (_Float16*)alloc((size_t)Ms * 64 * 2);
  _Float16* xcg_row  = (_Float16*)alloc((size_t)Mc * 64 * 2);
  _Float16* xcg_pack = (_Float16*)alloc((size_t)Mc * 64 * 2);
  _Float16* S_row  = (_Float16*)alloc((size_t)Ms * 64 * 2);
  _Float16* Sc_row = (_Float16*)alloc((size_t)Mc * 64 * 2);
  _Float16* Xgru  = (_Float16*)alloc((size_t)Ms * 96 * 2);
  _Float16* Xcgru = (_Float16*)alloc((size_t)Mc * 96 * 2);
  float* Gi_s = (float*)alloc((size_t)Ms * 192 * 4);
  float* Gh_s = (float*)alloc((size_t)Ms * 192 * 4);
  float* Gi_c = (float*)alloc((size_t)Mc * 192 * 4);
  float* Gh_c = (float*)alloc((size_t)Mc * 192 * 4);
  float*    H_s   = (float*)alloc((size_t)Ms * 64 * 4);
  _Float16* H16_s = (_Float16*)alloc((size_t)Ms * 64 * 2);
  float*    H_c   = (float*)alloc((size_t)Mc * 64 * 4);
  _Float16* H16_c = (_Float16*)alloc((size_t)Mc * 64 * 2);
  float* y_s = (float*)alloc(Ms * 4);
  float* y_c = (float*)alloc(Mc * 4);

  // ---- one-time setup ----
  deg_dinv<<<(Ns + 255) / 256, 256, 0, stream>>>(adj_sta, dinv_s, Ns);
  lap_build<<<(int)(((size_t)Ns * Ns + 255) / 256), 256, 0, stream>>>(adj_sta, dinv_s, Ls, Ns);
  deg_dinv<<<1, 256, 0, stream>>>(adj_city, dinv_c, Nc);
  lap_build<<<(Nc * Nc + 255) / 256, 256, 0, stream>>>(adj_city, dinv_c, Lc, Nc);
  afc_conv<<<(int)(((size_t)Ns * Nc + 255) / 256), 256, 0, stream>>>(afc, afc16, afcT16, Ns, Nc);

  auto packw = [&](const float* src, _Float16* dst, int Ksrc, int N) {
    int ntiles = N / 16, tiles = ((Ksrc + 31) / 32) * ntiles;
    pack_w<<<(tiles * 32 + 127) / 128, 128, 0, stream>>>(src, dst, Ksrc, ntiles, tiles, N);
  };
  packw(conv_w, W0sp, 28, 64);  packw(conv_w + 28 * 64, W1sp, 28, 64);
  packw(cconv_w, W0cp, 28, 64); packw(cconv_w + 28 * 64, W1cp, 28, 64);
  packw(gru_wi, wisp, 92, 192); packw(gru_wh, whsp, 64, 192);
  packw(cgru_wi, wicp, 92, 192); packw(cgru_wh, whcp, 64, 192);
  packw(c2s_w, c2sp, 64, 64);   packw(s2c_w, s2cp, 64, 64);

  hipMemsetAsync(H_s, 0, (size_t)Ms * 64 * 4, stream);
  hipMemsetAsync(H16_s, 0, (size_t)Ms * 64 * 2, stream);
  hipMemsetAsync(H_c, 0, (size_t)Mc * 64 * 4, stream);
  hipMemsetAsync(H16_c, 0, (size_t)Mc * 64 * 2, stream);

  // ---- 31 sequential steps ----
  for (int step = 0; step < Th - 1 + Tp; ++step) {
    const bool pred = step >= Th - 1;
    const int idx = step + 1;
    if (step == Th - 1) {
      init_yprev<<<(Ms + 255) / 256, 256, 0, stream>>>(x_hist, y_s, Ns, Th, Ms);
      init_yprev<<<(Mc + 255) / 256, 256, 0, stream>>>(cx_hist, y_c, Nc, Th, Mc);
    }
    const int f_t = pred ? Th + (step - (Th - 1)) : step + 1;

    build_x<<<(Ms * 32 + 255) / 256, 256, 0, stream>>>(
        x_hist, features, y_s, Xrow_s, Xpack_s, Xgru, Ns, step, f_t, pred, T, Dm, Th, Ms * 32);
    build_x<<<(Mc * 32 + 255) / 256, 256, 0, stream>>>(
        cx_hist, cfeats, y_c, Xrow_c, Xpack_c, Xcgru, Nc, step, f_t, pred, T, Dm, Th, Mc * 32);

    // T1 = L @ X  (per batch, N=32 -> NT=2, whole N per wave)
    gemm_wmma<0, 2><<<dim3(Ns / 128, 1, B), 256, 0, stream>>>(
        Ls, 0, Ns, Xpack_s, (long long)Ns * 32, nullptr, nullptr, 0, 0,
        T1s, (long long)Ns * 32, 32, 0, Ns, 32, Ns);
    gemm_wmma<0, 2><<<dim3(Nc / 128, 1, B), 256, 0, stream>>>(
        Lc, 0, Nc, Xpack_c, (long long)Nc * 32, nullptr, nullptr, 0, 0,
        T1c, (long long)Nc * 32, 32, 0, Nc, 32, Nc);

    // xg = sigmoid(X@W0 + T1@W1 + b)
    cheb_wmma<<<dim3(Ms / 128, 1), 256, 0, stream>>>(
        Xrow_s, 32, W0sp, T1s, 32, W1sp, conv_b, xg_row, 64, Ms);
    cheb_wmma<<<dim3(Mc / 128, 1), 256, 0, stream>>>(
        Xrow_c, 32, W0cp, T1c, 32, W1cp, cconv_b, xcg_row, 64, Mc);

    // pack xg / xcg as B-operands (per batch)
    {
      int tiles_s = (Ns / 32) * 4;  // K=Ns, N=64
      pack_b16<<<dim3((tiles_s * 32 + 127) / 128, 1, B), 128, 0, stream>>>(
          xg_row, (long long)Ns * 64, 64, xg_pack, (long long)Ns * 64, 4, tiles_s);
      int tiles_c = (Nc / 32) * 4;
      pack_b16<<<dim3((tiles_c * 32 + 127) / 128, 1, B), 128, 0, stream>>>(
          xcg_row, (long long)Nc * 64, 64, xcg_pack, (long long)Nc * 64, 4, tiles_c);
    }

    // S = afc @ xcg ; Sc = afc^T @ xg (original xg).  N=64 -> NT=4.
    gemm_wmma<0, 4><<<dim3(Ns / 128, 1, B), 256, 0, stream>>>(
        afc16, 0, Nc, xcg_pack, (long long)Nc * 64, nullptr, nullptr, 0, 0,
        S_row, (long long)Ns * 64, 64, 0, Ns, 64, Nc);
    gemm_wmma<0, 4><<<dim3(Nc / 128, 1, B), 256, 0, stream>>>(
        afcT16, 0, Ns, xg_pack, (long long)Ns * 64, nullptr, nullptr, 0, 0,
        Sc_row, (long long)Nc * 64, 64, 0, Nc, 64, Ns);

    // fused xg' = xg + S@c2s + b  -> directly into Xgru cols 28..91
    gemm_wmma<2, 4><<<dim3(Ms / 128, 1, 1), 256, 0, stream>>>(
        S_row, 0, 64, c2sp, 0, c2s_b, xg_row, 0, 64,
        Xgru, 0, 96, 28, Ms, 64, 64);
    gemm_wmma<2, 4><<<dim3(Mc / 128, 1, 1), 256, 0, stream>>>(
        Sc_row, 0, 64, s2cp, 0, s2c_b, xcg_row, 0, 64,
        Xcgru, 0, 96, 28, Mc, 64, 64);

    // GRU gates (N=192 -> 3 n-tile groups of NT=4)
    gemm_wmma<3, 4><<<dim3(Ms / 128, 3, 1), 256, 0, stream>>>(
        Xgru, 0, 96, wisp, 0, gru_bi, nullptr, 0, 0,
        Gi_s, 0, 192, 0, Ms, 192, 96);
    gemm_wmma<3, 4><<<dim3(Ms / 128, 3, 1), 256, 0, stream>>>(
        H16_s, 0, 64, whsp, 0, gru_bh, nullptr, 0, 0,
        Gh_s, 0, 192, 0, Ms, 192, 64);
    gemm_wmma<3, 4><<<dim3(Mc / 128, 3, 1), 256, 0, stream>>>(
        Xcgru, 0, 96, wicp, 0, cgru_bi, nullptr, 0, 0,
        Gi_c, 0, 192, 0, Mc, 192, 96);
    gemm_wmma<3, 4><<<dim3(Mc / 128, 3, 1), 256, 0, stream>>>(
        H16_c, 0, 64, whcp, 0, cgru_bh, nullptr, 0, 0,
        Gh_c, 0, 192, 0, Mc, 192, 64);

    gru_update<<<(Ms * 64 + 255) / 256, 256, 0, stream>>>(Gi_s, Gh_s, H_s, H16_s, Ms, idx);
    gru_update<<<(Mc * 64 + 255) / 256, 256, 0, stream>>>(Gi_c, Gh_c, H_c, H16_c, Mc, idx);

    if (pred) {
      const int t = step - (Th - 1);
      fc_out<<<(Ms + 255) / 256, 256, 0, stream>>>(H_s, fc_w, fc_b, y_s, out_s, Ns, t, Tp, Ms);
      fc_out<<<(Mc + 255) / 256, 256, 0, stream>>>(H_c, cfc_w, cfc_b, y_c, out_c, Nc, t, Tp, Mc);
    }
  }
}
